// ActivatedAttention_51445118272237
// MI455X (gfx1250) — matrix operand order repaired and checked by reading, not compiled, not run on hardware
//
#include <hip/hip_runtime.h>

// Problem constants (match reference)
constexpr int kB  = 4;
constexpr int kT  = 2048;
constexpr int kD  = 1024;
constexpr int kN3 = 3072;
constexpr float kLogTheta = 9.210340371976184f;   // ln(10000)

// Padded LDS tile strides (bank-conflict-free: gcd(20 dwords, 64 banks) -> all
// 16 lane start-banks distinct for 16B fragment loads).
constexpr int kLds = 40;   // bf16 elements per tile row (32 data + 8 pad)
constexpr int kLdc = 65;   // floats per C-stage row (64 data + 1 pad)

typedef __attribute__((ext_vector_type(16))) __bf16 v16bf;
typedef __attribute__((ext_vector_type(8)))  float  v8f;
typedef int v4i_ __attribute__((ext_vector_type(4)));
typedef int v8i_ __attribute__((ext_vector_type(8)));
typedef unsigned int v4u_ __attribute__((ext_vector_type(4)));
typedef __attribute__((address_space(1))) v4i_* as1_v4i;
typedef __attribute__((address_space(3))) v4i_* as3_v4i;

// ---------------------------------------------------------------------------
// CDNA5 async global->LDS copies (ASYNCcnt path), with 128-bit sync fallback.
// ---------------------------------------------------------------------------
#if defined(__has_builtin)
#if __has_builtin(__builtin_amdgcn_global_load_async_to_lds_b128) && \
    __has_builtin(__builtin_amdgcn_s_wait_asynccnt)
#define USE_ASYNC_LDS 1
#endif
#if __has_builtin(__builtin_amdgcn_tensor_load_to_lds) && \
    __has_builtin(__builtin_amdgcn_s_wait_tensorcnt)
#define USE_TDM 1
#endif
#endif
#ifndef USE_ASYNC_LDS
#define USE_ASYNC_LDS 0
#endif
#ifndef USE_TDM
#define USE_TDM 0
#endif

#if USE_ASYNC_LDS
static __device__ __forceinline__ void copy16_to_lds(const void* g, void* l) {
  as1_v4i gp = (as1_v4i)(unsigned long long)(__SIZE_TYPE__)g;
  as3_v4i lp = (as3_v4i)(unsigned int)(__SIZE_TYPE__)l;
  __builtin_amdgcn_global_load_async_to_lds_b128(gp, lp, 0, 0);
}
static __device__ __forceinline__ void copy_wait() {
  __builtin_amdgcn_s_wait_asynccnt(0);
}
#else
static __device__ __forceinline__ void copy16_to_lds(const void* g, void* l) {
  *(uint4*)l = *(const uint4*)g;
}
static __device__ __forceinline__ void copy_wait() {}
#endif

// ---------------------------------------------------------------------------
// Tensor Data Mover: one 2D tile DMA (128 rows x 32 bf16, row stride = Kdim)
// per operand, with D# LDS padding of 4 DWORDs every 16 DWORDs -> 80B LDS row
// stride (== kLds elements). D# layout per CDNA5 ISA 8.3/8.4.
// Toolchain builtin arity: 6 args (g0, g1, g2, g3, extra v8i, cpol).
// ---------------------------------------------------------------------------
#if USE_TDM
static __device__ __forceinline__ void tdm_load_tile_128x32_bf16(
    const void* gaddr, void* lds, int row_stride_elts) {
  const unsigned long long ga = (unsigned long long)(__SIZE_TYPE__)gaddr;
  const unsigned la = (unsigned)(__SIZE_TYPE__)lds;
  v4u_ g0;
  g0.x = 1u;                                             // count=1 (user D#)
  g0.y = la;                                             // lds_addr
  g0.z = (unsigned)ga;                                   // global_addr[31:0]
  g0.w = (unsigned)((ga >> 32) & 0x01FFFFFFull)          // global_addr[56:32]
         | (2u << 30);                                   // type=2 ("image")
  v8i_ g1;
  g1[0] = (1 << 16)                                      // data_size=2B
        | (1 << 20)                                      // pad_enable
        | (3 << 22)                                      // pad_interval: 16 DW
        | (3 << 25);                                     // pad_amount: 4 DW
  g1[1] = (int)(32u << 16);          // tensor_dim0[15:0]=32
  g1[2] = (int)(128u << 16);         // tensor_dim0[31:16]=0, tensor_dim1=128
  g1[3] = (int)(32u << 16);          // tensor_dim1 hi=0, tile_dim0=32
  g1[4] = 128;                       // tile_dim1=128, tile_dim2=0
  g1[5] = row_stride_elts;           // tensor_dim0_stride[31:0]
  g1[6] = 0;                         // stride hi, tensor_dim1_stride lo
  g1[7] = 0;
  const v4i_ z4 = {0, 0, 0, 0};
  const v8i_ z8 = {0, 0, 0, 0, 0, 0, 0, 0};
  __builtin_amdgcn_tensor_load_to_lds(g0, g1, z4, z4, z8, 0);
}
static __device__ __forceinline__ void tdm_wait() {
  __builtin_amdgcn_s_wait_tensorcnt(0);
}
#endif

// ---------------------------------------------------------------------------
// WMMA fragment gathers from "[index][k]" LDS tiles (stride kLds): every
// fragment is two contiguous 16B ds_load_b128 per lane, conflict-free.
//
// A (16x32 MxK): lanes 0-15: row=lane,    K={0..7,16..23}
//                lanes 16-31: row=lane-16, K={8..15,24..31}
// B (32x16 KxN): lanes 0-15: col=lane,    K={0..7,16..23}
//                lanes 16-31: col=lane-16, K={8..15,24..31}
// ---------------------------------------------------------------------------
static __device__ __forceinline__ v16bf load_a_frag(const __bf16* S, int rowb,
                                                    int lane) {
  const __bf16* p = S + (rowb + (lane & 15)) * kLds + ((lane < 16) ? 0 : 8);
  v16bf a;
#pragma unroll
  for (int j = 0; j < 8; ++j) a[j] = p[j];
#pragma unroll
  for (int j = 0; j < 8; ++j) a[8 + j] = p[16 + j];
  return a;
}

static __device__ __forceinline__ v16bf load_b_frag(const __bf16* S, int colb,
                                                    int lane) {
  const __bf16* p = S + (colb + (lane & 15)) * kLds + ((lane < 16) ? 0 : 8);
  v16bf b;
#pragma unroll
  for (int j = 0; j < 8; ++j) b[j] = p[j];
#pragma unroll
  for (int j = 0; j < 8; ++j) b[8 + j] = p[16 + j];
  return b;
}

// ===========================================================================
// Kernel 1: qkv = x @ W_in + b_in ; Q = relu(rope()), row-major bf16
//           K = relu(rope()), V = relu() -> written TRANSPOSED [b][d][t] bf16
// M=8192, N=3072, K=1024. Block 128x64, 8 waves x (2x2) WMMA tiles.
// Register-pipelined double-buffered LDS staging (fp32 -> bf16 convert).
// ===========================================================================
__global__ __launch_bounds__(256) void qkv_rope_kernel(
    const float* __restrict__ x, const float* __restrict__ W,
    const float* __restrict__ bias, __bf16* __restrict__ Qb,
    __bf16* __restrict__ Kt, __bf16* __restrict__ Vt) {
  __shared__ __bf16 As[2][128 * kLds];   // [row][k]
  __shared__ __bf16 Bs[2][64 * kLds];    // [col][k] (column-major stage)
  __shared__ float  Cs[128 * kLdc];

  const int tid  = threadIdx.x;
  const int lane = tid & 31;
  const int wave = tid >> 5;
  const int wm   = wave & 3;
  const int wn   = wave >> 2;
  const int m0   = blockIdx.x * 128;
  const int n0   = blockIdx.y * 64;

  const int ar = tid >> 1, ac = (tid & 1) * 16;    // A: 16 floats / thread
  const int bk = tid >> 3, bn = (tid & 7) * 8;     // B: 8 floats / thread

  float4 ra[4];
  float4 rb[2];

  auto load_regs = [&](int k0) {
    const float* ga = x + (size_t)(m0 + ar) * kD + k0 + ac;
#pragma unroll
    for (int u = 0; u < 4; ++u) ra[u] = *(const float4*)(ga + 4 * u);
    const float* gb = W + (size_t)(k0 + bk) * kN3 + n0 + bn;
#pragma unroll
    for (int u = 0; u < 2; ++u) rb[u] = *(const float4*)(gb + 4 * u);
    __builtin_prefetch(ga + 32, 0, 0);             // next K tile
    __builtin_prefetch(gb + (size_t)32 * kN3, 0, 0);
  };
  auto store_lds = [&](int buf) {
    __bf16* da = &As[buf][ar * kLds + ac];
    const float* fa = (const float*)ra;
#pragma unroll
    for (int j = 0; j < 16; ++j) da[j] = (__bf16)fa[j];
    const float* fb = (const float*)rb;
#pragma unroll
    for (int j = 0; j < 8; ++j) Bs[buf][(bn + j) * kLds + bk] = (__bf16)fb[j];
  };

  v8f acc[2][2];
#pragma unroll
  for (int i = 0; i < 2; ++i)
#pragma unroll
    for (int j = 0; j < 2; ++j)
#pragma unroll
      for (int r = 0; r < 8; ++r) acc[i][j][r] = 0.0f;

  load_regs(0);
  store_lds(0);
  __syncthreads();

  constexpr int KT = kD / 32;
  for (int kt = 0; kt < KT; ++kt) {
    const int  cur  = kt & 1;
    const bool more = (kt + 1 < KT);
    if (more) load_regs((kt + 1) * 32);

    v16bf af[2], bfr[2];
#pragma unroll
    for (int i = 0; i < 2; ++i) af[i]  = load_a_frag(&As[cur][0], wm * 32 + i * 16, lane);
#pragma unroll
    for (int j = 0; j < 2; ++j) bfr[j] = load_b_frag(&Bs[cur][0], wn * 32 + j * 16, lane);
#pragma unroll
    for (int i = 0; i < 2; ++i)
#pragma unroll
      for (int j = 0; j < 2; ++j)
        acc[i][j] = __builtin_amdgcn_wmma_f32_16x16x32_bf16(
            false, af[i], false, bfr[j], (short)0, acc[i][j], false, false);

    if (more) store_lds(cur ^ 1);
    __syncthreads();
  }

  // Stage C to LDS for the activation epilogue.
#pragma unroll
  for (int i = 0; i < 2; ++i)
#pragma unroll
    for (int j = 0; j < 2; ++j) {
      const int rbase = wm * 32 + i * 16 + ((lane >> 4) << 3);
      const int c     = wn * 32 + j * 16 + (lane & 15);
#pragma unroll
      for (int r = 0; r < 8; ++r) Cs[(rbase + r) * kLdc + c] = acc[i][j][r];
    }
  __syncthreads();

  const int seg    = n0 / kD;    // 0=Q, 1=K, 2=V (uniform: 64 | 1024)
  const int chbase = n0 % kD;

  if (seg == 0) {
    // Q: rope+relu, row-major write (pairs of channels per thread).
#pragma unroll
    for (int it = 0; it < 16; ++it) {
      const int pidx = it * 256 + tid;   // 128 rows x 32 pairs
      const int row  = pidx >> 5;
      const int pc   = pidx & 31;
      const int gm   = m0 + row;
      const int ch   = chbase + 2 * pc;
      const float v0 = Cs[row * kLdc + 2 * pc]     + bias[n0 + 2 * pc];
      const float v1 = Cs[row * kLdc + 2 * pc + 1] + bias[n0 + 2 * pc + 1];
      const float t   = (float)(gm & (kT - 1));
      const float inv = __expf(-((float)ch * (1.0f / kD)) * kLogTheta);
      const float ang = t * inv;
      const float cs = __cosf(ang), sn = __sinf(ang);
      Qb[(size_t)gm * kD + ch]     = (__bf16)fmaxf(v0 * cs - v1 * sn, 0.0f);
      Qb[(size_t)gm * kD + ch + 1] = (__bf16)fmaxf(v0 * sn + v1 * cs, 0.0f);
    }
  } else {
    // K/V: write transposed [b][ch][t]; each thread owns one channel and a
    // contiguous run of 32 t's -> contiguous 64B global writes.
    const int col = tid & 63;            // local channel
    const int t0  = (tid >> 6) * 32;     // row group
    const int ch  = chbase + col;
    const int bb  = m0 >> 11;            // batch (block never straddles)
    const int tb  = m0 & (kT - 1);
    __bf16* dstT = ((seg == 1) ? Kt : Vt) +
                   (size_t)bb * kD * kT + (size_t)ch * kT + tb + t0;
    const float bch = bias[n0 + col];
    if (seg == 1) {
      const float bpair = bias[n0 + (col ^ 1)];
      const float inv = __expf(-((float)(ch & ~1) * (1.0f / kD)) * kLogTheta);
      const bool odd = (ch & 1);
#pragma unroll
      for (int u = 0; u < 32; ++u) {
        const int row = t0 + u;
        const float vs = Cs[row * kLdc + col] + bch;
        const float vp = Cs[row * kLdc + (col ^ 1)] + bpair;
        const float ang = (float)(tb + row) * inv;
        const float cs = __cosf(ang), sn = __sinf(ang);
        const float r = odd ? (vp * sn + vs * cs) : (vs * cs - vp * sn);
        dstT[u] = (__bf16)fmaxf(r, 0.0f);
      }
    } else {
#pragma unroll
      for (int u = 0; u < 32; ++u)
        dstT[u] = (__bf16)fmaxf(Cs[(t0 + u) * kLdc + col] + bch, 0.0f);
    }
  }
}

// ===========================================================================
// Templated NT GEMM: C[m][n] = sum_k A[m][k] * B[n][k]  (both row-major-in-k,
// row stride == Kdim), C row-major with stride 1024. Block 128x128, 8 waves
// x (2x4) WMMA tiles. Double-buffered staging via TDM tile DMA (preferred),
// else per-lane async b128 copies, else plain 128-bit copies.
//   - kernel 2:  St[d2][d1] = sum_t Vt[d2][t] * Kt[d1][t]   (Sᵀ = Vᵀ K)
//   - kernel 3:  y[t][d2]   = sum_d1 Qb[t][d1] * St[d2][d1] (y  = Q S)
// ===========================================================================
template <typename OutT>
__global__ __launch_bounds__(256) void gemm_nt_kernel(
    const __bf16* __restrict__ A, size_t aBatch,
    const __bf16* __restrict__ Bt, size_t bBatch,
    OutT* __restrict__ C, size_t cBatch, int Kdim) {
  __shared__ __bf16 As[2][128 * kLds];   // [m][k]
  __shared__ __bf16 Bs[2][128 * kLds];   // [n][k]

  const int tid = threadIdx.x, lane = tid & 31, wave = tid >> 5;
  const int wm = wave & 3;             // 4 waves over 128 rows
  const int wn = wave >> 2;            // 2 waves over 128 cols (4 tiles each)
  const int m0 = blockIdx.x * 128, n0 = blockIdx.y * 128, bz = blockIdx.z;

  const __bf16* Ab = A + aBatch * bz;
  const __bf16* Bb = Bt + bBatch * bz;

#if USE_TDM
  auto stage = [&](int k0, int buf) {
    if (wave == 0) {   // one descriptor-driven DMA per operand tile
      tdm_load_tile_128x32_bf16(Ab + (size_t)m0 * Kdim + k0, &As[buf][0], Kdim);
      tdm_load_tile_128x32_bf16(Bb + (size_t)n0 * Kdim + k0, &Bs[buf][0], Kdim);
    }
  };
  auto stage_wait = [&]() {
    if (wave == 0) tdm_wait();
  };
#else
  const int sr = tid >> 1, sc = (tid & 1) * 16;   // 32B per thread per tile
  auto stage = [&](int k0, int buf) {
    const __bf16* ga = Ab + (size_t)(m0 + sr) * Kdim + k0 + sc;
    copy16_to_lds(ga,     &As[buf][sr * kLds + sc]);
    copy16_to_lds(ga + 8, &As[buf][sr * kLds + sc + 8]);
    const __bf16* gb = Bb + (size_t)(n0 + sr) * Kdim + k0 + sc;
    copy16_to_lds(gb,     &Bs[buf][sr * kLds + sc]);
    copy16_to_lds(gb + 8, &Bs[buf][sr * kLds + sc + 8]);
    __builtin_prefetch(ga + 32, 0, 0);
    __builtin_prefetch(gb + 32, 0, 0);
  };
  auto stage_wait = [&]() { copy_wait(); };
#endif

  v8f acc[2][4];
#pragma unroll
  for (int i = 0; i < 2; ++i)
#pragma unroll
    for (int j = 0; j < 4; ++j)
#pragma unroll
      for (int r = 0; r < 8; ++r) acc[i][j][r] = 0.0f;

  stage(0, 0);
  stage_wait();
  __syncthreads();

  const int KT = Kdim / 32;
  for (int kt = 0; kt < KT; ++kt) {
    const int cur = kt & 1;
    if (kt + 1 < KT) stage((kt + 1) * 32, cur ^ 1);

    v16bf af[2], bfr[4];
#pragma unroll
    for (int i = 0; i < 2; ++i) af[i]  = load_a_frag(&As[cur][0], wm * 32 + i * 16, lane);
#pragma unroll
    for (int j = 0; j < 4; ++j) bfr[j] = load_b_frag(&Bs[cur][0], wn * 64 + j * 16, lane);
#pragma unroll
    for (int i = 0; i < 2; ++i)
#pragma unroll
      for (int j = 0; j < 4; ++j)
        acc[i][j] = __builtin_amdgcn_wmma_f32_16x16x32_bf16(
            false, af[i], false, bfr[j], (short)0, acc[i][j], false, false);

    stage_wait();
    __syncthreads();
  }

  OutT* Cb = C + cBatch * bz;
#pragma unroll
  for (int i = 0; i < 2; ++i)
#pragma unroll
    for (int j = 0; j < 4; ++j) {
      const int rbase = m0 + wm * 32 + i * 16 + ((lane >> 4) << 3);
      const int gn    = n0 + wn * 64 + j * 16 + (lane & 15);
#pragma unroll
      for (int r = 0; r < 8; ++r)
        Cb[(size_t)(rbase + r) * kD + gn] = (OutT)acc[i][j][r];
    }
}

// ===========================================================================
// Kernel 4: GroupNorm (32 groups x 32 ch) per row of y, then the final
// swapaxes(-2,-1) write: out[b, c, t] = yn[b, t, c].
// ===========================================================================
__global__ __launch_bounds__(256) void gn_transpose_kernel(
    const float* __restrict__ y, const float* __restrict__ gw,
    const float* __restrict__ gb, float* __restrict__ out) {
  const int row = blockIdx.x;
  const int tid = threadIdx.x;
  const float* yr = y + (size_t)row * kD;
  const int c0 = tid * 4;

  const float4 v = *(const float4*)(yr + c0);
  float s  = v.x + v.y + v.z + v.w;
  float ss = v.x * v.x + v.y * v.y + v.z * v.z + v.w * v.w;
#pragma unroll
  for (int m = 1; m <= 4; m <<= 1) {
    s  += __shfl_xor(s,  m, 32);
    ss += __shfl_xor(ss, m, 32);
  }
  const float mean = s * (1.0f / 32.0f);
  const float var  = ss * (1.0f / 32.0f) - mean * mean;
  const float rs   = rsqrtf(var + 1e-5f);

  const int b = row >> 11;
  const int t = row & (kT - 1);
  float* ob = out + (size_t)b * kD * kT + t;
  const float vals[4] = {v.x, v.y, v.z, v.w};
#pragma unroll
  for (int u = 0; u < 4; ++u) {
    const int c = c0 + u;
    ob[(size_t)c * kT] = (vals[u] - mean) * rs * gw[c] + gb[c];
  }
}

// ===========================================================================
extern "C" void kernel_launch(void* const* d_in, const int* in_sizes, int n_in,
                              void* d_out, int out_size, void* d_ws,
                              size_t ws_size, hipStream_t stream) {
  const float* x   = (const float*)d_in[0];
  const float* W   = (const float*)d_in[1];
  const float* bin = (const float*)d_in[2];
  const float* gw  = (const float*)d_in[3];
  const float* gb  = (const float*)d_in[4];
  float* out = (float*)d_out;

  char* ws = (char*)d_ws;
  const size_t szQKV = (size_t)kB * kT * kD * 2;   // bf16, 16 MB each
  const size_t szS   = (size_t)kB * kD * kD * 2;   // bf16, 8 MB
  __bf16* Qb = (__bf16*)(ws);                      // [b][t][d]
  __bf16* Kt = (__bf16*)(ws + szQKV);              // [b][d][t]
  __bf16* Vt = (__bf16*)(ws + 2 * szQKV);          // [b][d][t]
  __bf16* St = (__bf16*)(ws + 3 * szQKV);          // [b][d2][d1]  (S^T)
  float*  y  = (float*)(ws + 3 * szQKV + szS);     // [b][t][d] fp32
  // workspace total: 88 MB

  qkv_rope_kernel<<<dim3((kB * kT) / 128, kN3 / 64), 256, 0, stream>>>(
      x, W, bin, Qb, Kt, Vt);

  // St[d2][d1] = sum_t Vt[d2][t] * Kt[d1][t]
  gemm_nt_kernel<__bf16><<<dim3(kD / 128, kD / 128, kB), 256, 0, stream>>>(
      Vt, (size_t)kD * kT, Kt, (size_t)kD * kT, St, (size_t)kD * kD, kT);

  // y[t][d2] = sum_d1 Qb[t][d1] * St[d2][d1]
  gemm_nt_kernel<float><<<dim3(kT / 128, kD / 128, kB), 256, 0, stream>>>(
      Qb, (size_t)kT * kD, St, (size_t)kD * kD, y, (size_t)kT * kD, kD);

  gn_transpose_kernel<<<dim3(kB * kT), 256, 0, stream>>>(y, gw, gb, out);
}
